// MHA_86122684219585
// MI455X (gfx1250) — matrix operand order repaired and checked by reading, not compiled, not run on hardware
//
#include <hip/hip_runtime.h>
#include <stdint.h>

#define B_ 2
#define S_ 2048
#define H_ 16
#define D_ 64
#define C2f 0.18033688011112042f   // D^-0.5 * log2(e)
#define BIGC 1024.0f               // masked-score sentinel (|real scores| << 1024)
#define TK 64                      // keys per tile
#define NT (S_ / TK)               // 32 tiles
#define KP 72                      // LDS row pitch in ushorts (64 + 8 pad -> conflict-free)

typedef __attribute__((ext_vector_type(16))) __bf16   v16bf;
typedef __attribute__((ext_vector_type(8)))  float    v8f;
typedef __attribute__((ext_vector_type(4)))  unsigned v4u;
typedef __attribute__((ext_vector_type(8)))  unsigned v8u;

// two f32 -> packed bf16 dword: one v_perm_b32 (+ cheap round bias)
__device__ __forceinline__ unsigned pack_bf16(float lo, float hi) {
    unsigned a = __builtin_bit_cast(unsigned, lo) + 0x8000u;
    unsigned b = __builtin_bit_cast(unsigned, hi) + 0x8000u;
    return __builtin_amdgcn_perm(b, a, 0x07060302u);   // {b[31:16], a[31:16]}
}

// bool byte -> 0.0f / 1.0f; backend combines uitofp(and(lshr)) into v_cvt_f32_ubyteN
__device__ __forceinline__ float mbyte(unsigned w, int i) {
    return (float)((w >> (8 * i)) & 0xffu);
}

__device__ __forceinline__ v16bf ld_frag(const unsigned short* p) {
    v4u c0 = *(const v4u*)(p);        // 16B chunk (K/e = base .. base+7)
    v4u c1 = *(const v4u*)(p + 8);    // 16B chunk (K/e = base+8 .. +15)
    v8u a; a.lo = c0; a.hi = c1;
    return __builtin_bit_cast(v16bf, a);
}

__global__ __launch_bounds__(256, 2)
void MHA_flash_fwd(const float* __restrict__ qp, const float* __restrict__ kp,
                   const float* __restrict__ vp, const unsigned char* __restrict__ maskp,
                   float* __restrict__ outp)
{
    __shared__ __align__(16) unsigned short ldsK[2][TK * KP];  // [key][d]  bf16
    __shared__ __align__(16) unsigned short ldsV[2][64 * KP];  // [d][key]  bf16 (V^T)

    const int tid  = threadIdx.x;
    const int lane = tid & 31;
    const int wave = tid >> 5;
    const int l16  = lane & 15;
    const int hi   = lane >> 4;

    const int nQB  = S_ / 128;
    const int wg   = blockIdx.x;
    const int qblk = wg % nQB;
    const int h    = (wg / nQB) % H_;
    const int b    =  wg / (nQB * H_);

    const int q0  = qblk * 128 + wave * 16;
    const int myq = q0 + l16;

    // ---- Q fragments: B-matrix of S^T = K * Q^T (lane N=query, K(d)=e+16*hi+32*ds)
    const float* qrow = qp + (((size_t)(b * S_) + myq) * H_ + h) * D_;
    v16bf qf[2];
#pragma unroll
    for (int ds = 0; ds < 2; ++ds) {
        const int d0 = ds * 32 + hi * 16;
        v8u qu;
#pragma unroll
        for (int c = 0; c < 4; ++c) {
            float4 f = *(const float4*)(qrow + d0 + c * 4);
            qu[2 * c + 0] = pack_bf16(f.x, f.y);
            qu[2 * c + 1] = pack_bf16(f.z, f.w);
        }
        qf[ds] = __builtin_bit_cast(v16bf, qu);
    }

    const unsigned char* mrow = maskp + (((size_t)(b * S_) + myq) * H_ + h) * S_;

    // ------- cooperative staging (double buffered), all in registers first -------
    const int krow[4] = { (0 * 256 + tid) >> 4, (1 * 256 + tid) >> 4,
                          (2 * 256 + tid) >> 4, (3 * 256 + tid) >> 4 };
    const int kc4   = (tid & 15) * 4;
    const int vrow0 = (tid >> 4) * 4;
    const int vcol  = (tid & 15) * 4;

    float4 kreg[4], vreg[4];
    // prologue: stage tile 0 into buffer 0
#pragma unroll
    for (int it = 0; it < 4; ++it)
        kreg[it] = *(const float4*)(kp + (((size_t)(b * S_) + krow[it]) * H_ + h) * D_ + kc4);
#pragma unroll
    for (int i = 0; i < 4; ++i)
        vreg[i]  = *(const float4*)(vp + (((size_t)(b * S_) + vrow0 + i) * H_ + h) * D_ + vcol);
#pragma unroll
    for (int it = 0; it < 4; ++it) {
        uint2 pk2; pk2.x = pack_bf16(kreg[it].x, kreg[it].y);
                   pk2.y = pack_bf16(kreg[it].z, kreg[it].w);
        *(uint2*)&ldsK[0][krow[it] * KP + kc4] = pk2;
    }
#pragma unroll
    for (int j = 0; j < 4; ++j) {          // 4x4 register transpose for V^T
        uint2 pv2; pv2.x = pack_bf16((&vreg[0].x)[j], (&vreg[1].x)[j]);
                   pv2.y = pack_bf16((&vreg[2].x)[j], (&vreg[3].x)[j]);
        *(uint2*)&ldsV[0][(vcol + j) * KP + vrow0] = pv2;
    }

    float m = -1.0e30f, l = 0.0f;
    v8f o[4] = {};                        // O^T accum: lane N=query, M=d (4 slices)

    for (int t = 0; t < NT; ++t) {
        __syncthreads();                  // staging of tile t complete
        const int t0  = t * TK;
        const int buf = t & 1;
        const bool more = (t + 1 < NT);

        // ---- mask rows for this tile: issue early, consumed after QK^T ----
        uint2 mw[4];
#pragma unroll
        for (int kg = 0; kg < 4; ++kg)
            mw[kg] = *(const uint2*)(mrow + t0 + kg * 16 + hi * 8);

        // ---- prefetch next tile's K/V into registers (overlaps compute) ----
        if (more) {
            const int n0 = t0 + TK;
#pragma unroll
            for (int it = 0; it < 4; ++it)
                kreg[it] = *(const float4*)(kp + (((size_t)(b * S_) + n0 + krow[it]) * H_ + h) * D_ + kc4);
#pragma unroll
            for (int i = 0; i < 4; ++i)
                vreg[i]  = *(const float4*)(vp + (((size_t)(b * S_) + n0 + vrow0 + i) * H_ + h) * D_ + vcol);
        }

        // ---- S^T = K * Q^T : 4 independent chains, 2-deep fragment pipeline ----
        v8f s[4] = {};
        {
            v16bf af[2];
            af[0] = ld_frag(&ldsK[buf][l16 * KP + hi * 16]);       // step 0: kg0, ds0
#pragma unroll
            for (int i = 0; i < 8; ++i) {
                const int kg = i & 3, ds = i >> 2;
                if (i + 1 < 8) {
                    const int kg1 = (i + 1) & 3, ds1 = (i + 1) >> 2;
                    af[(i + 1) & 1] =
                        ld_frag(&ldsK[buf][(kg1 * 16 + l16) * KP + ds1 * 32 + hi * 16]);
                }
                s[kg] = __builtin_amdgcn_wmma_f32_16x16x32_bf16(
                            false, af[i & 1], false, qf[ds], (short)0, s[kg], false, false);
            }
        }

        // ---- mask + online softmax, exp2 domain, per-lane rows + one shfl ----
        float xm[4][8];
        float mloc = -1.0e30f;
#pragma unroll
        for (int kg = 0; kg < 4; ++kg)
#pragma unroll
            for (int r = 0; r < 8; ++r) {
                const float mf = (r < 4) ? mbyte(mw[kg].x, r) : mbyte(mw[kg].y, r - 4);
                const float x1 = __builtin_fmaf(s[kg][r], C2f, BIGC);
                const float xv = __builtin_fmaf(mf, x1, -BIGC);  // masked -> exactly -1024
                xm[kg][r] = xv;
                mloc = fmaxf(mloc, xv);
            }
        mloc = fmaxf(mloc, __shfl_xor(mloc, 16, 32));
        const float mnew  = fmaxf(m, mloc);
        const float alpha = __builtin_amdgcn_exp2f(m - mnew);
        float ssum = 0.0f;
#pragma unroll
        for (int kg = 0; kg < 4; ++kg)
#pragma unroll
            for (int r = 0; r < 8; ++r) {
                xm[kg][r] = __builtin_amdgcn_exp2f(xm[kg][r] - mnew);
                ssum += xm[kg][r];
            }
        l = l * alpha + (ssum + __shfl_xor(ssum, 16, 32));
        m = mnew;
#pragma unroll
        for (int ds = 0; ds < 4; ++ds) o[ds] *= alpha;     // v_pk_mul_f32

        // ---- rebuild P^T B-fragments (half-wave exchange) + pack to bf16 ----
        v16bf pf[2];
#pragma unroll
        for (int kh = 0; kh < 2; ++kh) {
            float v0[8], v1[8];
#pragma unroll
            for (int e = 0; e < 8; ++e) {
                const float tB = __shfl_xor(xm[kh * 2 + 1][e], 16, 32);
                const float tA = __shfl_xor(xm[kh * 2 + 0][e], 16, 32);
                v0[e] = hi ? tB : xm[kh * 2 + 0][e];   // K = e     (+16 on hi half)
                v1[e] = hi ? xm[kh * 2 + 1][e] : tA;   // K = 8 + e (+16 on hi half)
            }
            v8u pu;
#pragma unroll
            for (int j = 0; j < 4; ++j) {
                pu[j]     = pack_bf16(v0[2 * j], v0[2 * j + 1]);
                pu[4 + j] = pack_bf16(v1[2 * j], v1[2 * j + 1]);
            }
            pf[kh] = __builtin_bit_cast(v16bf, pu);
        }

        // ---- O^T += V^T * P^T : 4 independent chains, 2-deep fragment pipeline ----
        {
            v16bf af[2];
            af[0] = ld_frag(&ldsV[buf][l16 * KP + hi * 16]);       // step 0: ds0, kh0
#pragma unroll
            for (int i = 0; i < 8; ++i) {
                const int ds = i & 3, kh = i >> 2;
                if (i + 1 < 8) {
                    const int ds1 = (i + 1) & 3, kh1 = (i + 1) >> 2;
                    af[(i + 1) & 1] =
                        ld_frag(&ldsV[buf][(ds1 * 16 + l16) * KP + kh1 * 32 + hi * 16]);
                }
                o[ds] = __builtin_amdgcn_wmma_f32_16x16x32_bf16(
                            false, af[i & 1], false, pf[kh], (short)0, o[ds], false, false);
            }
        }

        // ---- store prefetched tile t+1 into the other buffer (no barrier) ----
        if (more) {
            const int nbuf = (t + 1) & 1;
#pragma unroll
            for (int it = 0; it < 4; ++it) {
                uint2 pk2; pk2.x = pack_bf16(kreg[it].x, kreg[it].y);
                           pk2.y = pack_bf16(kreg[it].z, kreg[it].w);
                *(uint2*)&ldsK[nbuf][krow[it] * KP + kc4] = pk2;
            }
#pragma unroll
            for (int j = 0; j < 4; ++j) {
                uint2 pv2; pv2.x = pack_bf16((&vreg[0].x)[j], (&vreg[1].x)[j]);
                           pv2.y = pack_bf16((&vreg[2].x)[j], (&vreg[3].x)[j]);
                *(uint2*)&ldsV[nbuf][(vcol + j) * KP + vrow0] = pv2;
            }
        }
    }

    // ---------------- epilogue: normalize, contiguous 32B stores ----------------
    const float invl = 1.0f / l;
    float* orow = outp + ((size_t)(b * S_) + myq) * (H_ * D_) + h * D_;
#pragma unroll
    for (int ds = 0; ds < 4; ++ds) {
        const v8f of = o[ds] * invl;
        float4 f0, f1;
        f0.x = of[0]; f0.y = of[1]; f0.z = of[2]; f0.w = of[3];
        f1.x = of[4]; f1.y = of[5]; f1.z = of[6]; f1.w = of[7];
        *(float4*)(orow + ds * 16 + hi * 8)     = f0;
        *(float4*)(orow + ds * 16 + hi * 8 + 4) = f1;
    }
}

extern "C" void kernel_launch(void* const* d_in, const int* in_sizes, int n_in,
                              void* d_out, int out_size, void* d_ws, size_t ws_size,
                              hipStream_t stream) {
    (void)in_sizes; (void)n_in; (void)out_size; (void)d_ws; (void)ws_size;
    const float* q = (const float*)d_in[0];
    const float* k = (const float*)d_in[1];
    const float* v = (const float*)d_in[2];
    const unsigned char* mask = (const unsigned char*)d_in[3];  // jnp.bool_ = 1 byte
    float* out = (float*)d_out;

    const int grid = B_ * H_ * (S_ / 128);   // 512 workgroups x 8 waves
    MHA_flash_fwd<<<dim3(grid), dim3(256), 0, stream>>>(q, k, v, mask, out);
}
